// ChamferDistance_8821862826558
// MI455X (gfx1250) — matrix operand order repaired and checked by reading, not compiled, not run on hardware
//
#include <hip/hip_runtime.h>
#include <hip/hip_bf16.h>

typedef __attribute__((ext_vector_type(16))) __bf16 v16bf;
typedef __attribute__((ext_vector_type(8)))  __bf16 v8bf;
typedef __attribute__((ext_vector_type(8)))  float  v8f;

constexpr int BB = 8;     // batches
constexpr int NN = 4096;  // points1 count
constexpr int MM = 4096;  // points2 count
constexpr int DD = 128;   // feature dim
constexpr int TN = 128;   // points1 rows per block (8 waves x 16 rows)
constexpr int TM = 64;    // points2 rows per block (4 WMMA col-tiles)
constexpr int LDSS = 144; // LDS row stride in bf16 elems (128 + 16 pad -> conflict-friendly)

// ---- order-preserving float<->uint encoding so atomicMin(u32) == float min ----
__device__ __forceinline__ unsigned enc_f32(float f) {
  unsigned u = __float_as_uint(f);
  return (u & 0x80000000u) ? ~u : (u | 0x80000000u);
}
__device__ __forceinline__ float dec_f32(unsigned u) {
  unsigned b = (u & 0x80000000u) ? (u & 0x7FFFFFFFu) : ~u;
  return __uint_as_float(b);
}

// split 8 fp32 values into bf16 hi + bf16 residual-lo fragments
__device__ __forceinline__ void cvt8_hilo(float4 x, float4 y, v16bf& h, v16bf& l, int base) {
  float f[8] = {x.x, x.y, x.z, x.w, y.x, y.y, y.z, y.w};
#pragma unroll
  for (int i = 0; i < 8; ++i) {
    __bf16 hv = (__bf16)f[i];
    h[base + i] = hv;
    l[base + i] = (__bf16)(f[i] - (float)hv);
  }
}

// ---------------- kernel 1: squared norms + min-buffer init ----------------
__global__ __launch_bounds__(256) void chamfer_prep(
    const float* __restrict__ p1, const float* __restrict__ p2,
    float* __restrict__ sq1, float* __restrict__ sq2,
    unsigned* __restrict__ m1, unsigned* __restrict__ m2) {
  int idx = blockIdx.x * blockDim.x + threadIdx.x;
  const int BN = BB * NN;
  const int BM = BB * MM;
  if (idx < BN) {
    const float4* r = (const float4*)(p1 + (size_t)idx * DD);
    float s = 0.f;
#pragma unroll
    for (int i = 0; i < DD / 4; ++i) {
      float4 v = r[i];
      s += v.x * v.x + v.y * v.y + v.z * v.z + v.w * v.w;
    }
    sq1[idx] = s;
    m1[idx] = 0xFFFFFFFFu;
  } else if (idx < BN + BM) {
    int j = idx - BN;
    const float4* r = (const float4*)(p2 + (size_t)j * DD);
    float s = 0.f;
#pragma unroll
    for (int i = 0; i < DD / 4; ++i) {
      float4 v = r[i];
      s += v.x * v.x + v.y * v.y + v.z * v.z + v.w * v.w;
    }
    sq2[j] = s;
    m2[j] = 0xFFFFFFFFu;
  }
}

// ---------------- kernel 2: fused WMMA cross-term + min reductions ----------------
__global__ __launch_bounds__(256) void chamfer_main(
    const float* __restrict__ p1, const float* __restrict__ p2,
    const float* __restrict__ sq1, const float* __restrict__ sq2,
    unsigned* __restrict__ gmin1, unsigned* __restrict__ gmin2) {
  __shared__ __bf16 ldsH[TM * LDSS];
  __shared__ __bf16 ldsL[TM * LDSS];
  __shared__ unsigned colminU[TM];

  const int m0 = blockIdx.x * TM;
  const int n0 = blockIdx.y * TN;
  const int b  = blockIdx.z;
  const int tid = threadIdx.x;

  if (tid < TM) colminU[tid] = 0xFFFFFFFFu;

  // Stage points2 tile [TM x DD] into LDS as bf16 hi/lo planes.
  {
    const int row = tid >> 2;            // 64 rows, 4 threads per row
    const int c4  = (tid & 3) * 8;       // 8 float4 per thread
    const float4* src = (const float4*)(p2 + ((size_t)b * MM + m0 + row) * DD);
#pragma unroll
    for (int j = 0; j < 8; ++j) {
      float4 v = src[c4 + j];
      const int d = (c4 + j) * 4;
      __bf16* dh = &ldsH[row * LDSS + d];
      __bf16* dl = &ldsL[row * LDSS + d];
      float f0 = v.x, f1 = v.y, f2 = v.z, f3 = v.w;
      __bf16 h0 = (__bf16)f0, h1 = (__bf16)f1, h2 = (__bf16)f2, h3 = (__bf16)f3;
      dh[0] = h0; dh[1] = h1; dh[2] = h2; dh[3] = h3;
      dl[0] = (__bf16)(f0 - (float)h0);
      dl[1] = (__bf16)(f1 - (float)h1);
      dl[2] = (__bf16)(f2 - (float)h2);
      dl[3] = (__bf16)(f3 - (float)h3);
    }
  }
  __syncthreads();

  const int wave = tid >> 5;
  const int lane = tid & 31;
  const int r  = lane & 15;   // A: row within 16-tile, B: col within 16-tile
  const int hs = lane >> 4;   // lane half select

  const float* arow = p1 + ((size_t)b * NN + n0 + wave * 16 + r) * DD;

  v8f acc[4];
#pragma unroll
  for (int mt = 0; mt < 4; ++mt) acc[mt] = (v8f){0.f, 0.f, 0.f, 0.f, 0.f, 0.f, 0.f, 0.f};

#pragma unroll
  for (int kc = 0; kc < 4; ++kc) {
    const int k0 = kc * 32;
    // A fragment (16-bit 16x32 layout): lane half hs -> K base offset 8*hs,
    // elems [d0a..d0a+7] and [d0a+16..d0a+23]
    const int d0a = k0 + hs * 8;
    const float4* pa0 = (const float4*)(arow + d0a);
    const float4* pa1 = (const float4*)(arow + d0a + 16);
    float4 va0 = pa0[0], va1 = pa0[1];
    float4 va2 = pa1[0], va3 = pa1[1];
    v16bf ah, al;
    cvt8_hilo(va0, va1, ah, al, 0);
    cvt8_hilo(va2, va3, ah, al, 8);

    // B fragments (16-bit 32x16 layout): lane half hs -> K [k0+8*hs..+7] and [k0+8*hs+16..+23].
    // Load ALL 8 fragments first so the ds_loads clause together and the 12 WMMAs
    // below can issue back-to-back.
    const int d0b = k0 + hs * 8;
    v16bf bh[4], bl[4];
#pragma unroll
    for (int mt = 0; mt < 4; ++mt) {
      const int c = mt * 16 + r;
      const v8bf* bhp = (const v8bf*)&ldsH[c * LDSS + d0b];
      const v8bf* blp = (const v8bf*)&ldsL[c * LDSS + d0b];
      bh[mt] = __builtin_shufflevector(bhp[0], bhp[2],
          0, 1, 2, 3, 4, 5, 6, 7, 8, 9, 10, 11, 12, 13, 14, 15);
      bl[mt] = __builtin_shufflevector(blp[0], blp[2],
          0, 1, 2, 3, 4, 5, 6, 7, 8, 9, 10, 11, 12, 13, 14, 15);
    }

    // Product-major issue order: dependent WMMAs on the same accumulator are
    // always 4 independent WMMAs apart, covering the 1+4-slot bf16 WMMA->WMMA
    // hazard window without v_nop padding.
#pragma unroll
    for (int mt = 0; mt < 4; ++mt)
      acc[mt] = __builtin_amdgcn_wmma_f32_16x16x32_bf16(false, ah, false, bh[mt],
                                                        (short)0, acc[mt], false, false);
#pragma unroll
    for (int mt = 0; mt < 4; ++mt)
      acc[mt] = __builtin_amdgcn_wmma_f32_16x16x32_bf16(false, ah, false, bl[mt],
                                                        (short)0, acc[mt], false, false);
#pragma unroll
    for (int mt = 0; mt < 4; ++mt)
      acc[mt] = __builtin_amdgcn_wmma_f32_16x16x32_bf16(false, al, false, bh[mt],
                                                        (short)0, acc[mt], false, false);
  }

  // Epilogue: dist = sq1[n] + sq2[m] - 2*cross; fused row/col min reductions.
  // C layout: VGPR vi holds row (wave*16 + hs*8 + vi), col = m0 + mt*16 + (lane&15).
  const float FMAX = 3.402823466e38f;
  float s1v[8];
#pragma unroll
  for (int vi = 0; vi < 8; ++vi)
    s1v[vi] = sq1[(size_t)b * NN + n0 + wave * 16 + hs * 8 + vi];

  float rowmin[8];
#pragma unroll
  for (int vi = 0; vi < 8; ++vi) rowmin[vi] = FMAX;

#pragma unroll
  for (int mt = 0; mt < 4; ++mt) {
    const int gm = m0 + mt * 16 + r;
    const float s2 = sq2[(size_t)b * MM + gm];
    float cmin = FMAX;
#pragma unroll
    for (int vi = 0; vi < 8; ++vi) {
      float dist = s1v[vi] + s2 - 2.0f * acc[mt][vi];
      rowmin[vi] = fminf(rowmin[vi], dist);
      cmin = fminf(cmin, dist);
    }
    // combine the two lane-halves (rows 0-7 vs 8-15 of this wave's strip)
    cmin = fminf(cmin, __shfl_xor(cmin, 16, 32));
    if (lane < 16) atomicMin(&colminU[mt * 16 + lane], enc_f32(cmin));
  }

#pragma unroll
  for (int vi = 0; vi < 8; ++vi) {
    float v = rowmin[vi];
    v = fminf(v, __shfl_xor(v, 1, 32));
    v = fminf(v, __shfl_xor(v, 2, 32));
    v = fminf(v, __shfl_xor(v, 4, 32));
    v = fminf(v, __shfl_xor(v, 8, 32));
    if (r == 0)
      atomicMin(&gmin1[(size_t)b * NN + n0 + wave * 16 + hs * 8 + vi], enc_f32(v));
  }

  __syncthreads();
  if (tid < TM) atomicMin(&gmin2[(size_t)b * MM + m0 + tid], colminU[tid]);
}

// ---------------- kernel 3: final mean(min1) + mean(min2) ----------------
__global__ __launch_bounds__(256) void chamfer_finalize(
    const unsigned* __restrict__ m1, const unsigned* __restrict__ m2,
    float* __restrict__ out) {
  __shared__ float red[256];
  const int BN = BB * NN;
  const int BM = BB * MM;
  float s1 = 0.f, s2 = 0.f;
  for (int i = threadIdx.x; i < BN; i += 256) s1 += dec_f32(m1[i]);
  for (int i = threadIdx.x; i < BM; i += 256) s2 += dec_f32(m2[i]);
  red[threadIdx.x] = s1 / (float)BN + s2 / (float)BM;
  __syncthreads();
  for (int off = 128; off > 0; off >>= 1) {
    if (threadIdx.x < off) red[threadIdx.x] += red[threadIdx.x + off];
    __syncthreads();
  }
  if (threadIdx.x == 0) out[0] = red[0];
}

extern "C" void kernel_launch(void* const* d_in, const int* in_sizes, int n_in,
                              void* d_out, int out_size, void* d_ws, size_t ws_size,
                              hipStream_t stream) {
  (void)in_sizes; (void)n_in; (void)out_size; (void)ws_size;
  const float* p1 = (const float*)d_in[0];  // [B, N, D] fp32
  const float* p2 = (const float*)d_in[1];  // [B, M, D] fp32

  // workspace layout (512 KB total): sq1 | sq2 | min1(enc u32) | min2(enc u32)
  float*    sq1 = (float*)d_ws;
  float*    sq2 = sq1 + (size_t)BB * NN;
  unsigned* m1  = (unsigned*)(sq2 + (size_t)BB * MM);
  unsigned* m2  = m1 + (size_t)BB * NN;

  const int total = BB * (NN + MM);
  chamfer_prep<<<(total + 255) / 256, 256, 0, stream>>>(p1, p2, sq1, sq2, m1, m2);

  dim3 grid(MM / TM, NN / TN, BB);
  chamfer_main<<<grid, 256, 0, stream>>>(p1, p2, sq1, sq2, m1, m2);

  chamfer_finalize<<<1, 256, 0, stream>>>(m1, m2, (float*)d_out);
}
